// AttSeqM_67748814127286
// MI455X (gfx1250) — compile-verified, matrix operand order checked by reading, not run on hardware
//
#include <hip/hip_runtime.h>
#include <hip/hip_bf16.h>

// ---------------------------------------------------------------------------
// AttSeqM fused attention for MI455X (gfx1250), wave32 + WMMA f16.
//
// Shapes: B=2048, S=200(pad 208), IN_F=128, H=8, QLEN=16, VLEN=64, HID=512.
// One workgroup (8 waves) per batch element; all intermediates live in LDS,
// V projection is computed 16-row-tile at a time and contracted against the
// softmax probabilities immediately (never spilled to HBM).
//
// Roofline: ~135 GFLOP total, ~200 MB mandatory HBM traffic (~9 us floor at
// 23.3 TB/s) -> compute-bound on the projections -> WMMA f16 with f32 acc.
// Fast rcp/rsq (v_rcp_f32 / v_rsq_f32) for sigmoid gates, softmax and LN to
// keep the VALU tail off the WMMA critical path.
// ---------------------------------------------------------------------------

typedef __attribute__((ext_vector_type(16))) _Float16 v16h;
typedef __attribute__((ext_vector_type(8)))  _Float16 v8h;
typedef __attribute__((ext_vector_type(8)))  float    v8f;

namespace asq {
constexpr int kB      = 2048;
constexpr int kS      = 200;
constexpr int kSPad   = 208;     // 13 * 16
constexpr int kMT     = 13;      // M tiles of 16 rows
constexpr int kInQ    = 120;
constexpr int kPosE   = 8;
constexpr int kInF    = 128;     // GEMM K dimension
constexpr int kHeads  = 8;
constexpr int kQLen   = 16;
constexpr int kVLen   = 64;
constexpr int kHid    = 512;
constexpr float kLnEps = 1e-5f;

// LDS strides chosen against bank conflicts:
//  - x rows: 136 halves (272B, keeps 16B alignment for ds_load_b128)
//  - k rows: 129 floats -> score-stage bank = (s + d) % 64, conflict-free
constexpr int kXStride = 136;
constexpr int kKStride = 129;

// workspace (f16 halves) layout for pre-swizzled weights
constexpr size_t kOffWk  = 0;
constexpr size_t kOffWkc = 128 * 128;
constexpr size_t kOffWv  = 2 * 128 * 128;
constexpr size_t kOffWvc = 2 * 128 * 128 + 128 * 512;

// dynamic LDS layout (bytes)
constexpr size_t kLdsXv    = 0;                                   // 208*136 f16
constexpr size_t kLdsK     = kLdsXv + kSPad * kXStride * 2;       // 208*129 f32
constexpr size_t kLdsProbs = kLdsK + kSPad * kKStride * 4;        // 8*208 f32
constexpr size_t kLdsQ     = kLdsProbs + kHeads * kSPad * 4;      // 128 f32
constexpr size_t kLdsRed   = kLdsQ + 128 * 4;                     // 16*16 f32
constexpr size_t kLdsMurs  = kLdsRed + 16 * 16 * 4;               // 32 f32
constexpr size_t kLdsBytes = kLdsMurs + 32 * 4;                   // ~173 KB
} // namespace asq

// fast sigmoid: v_exp_f32 + v_add + v_rcp_f32 (no IEEE div fixup sequence)
__device__ __forceinline__ float asq_sigmoid(float x) {
    return __builtin_amdgcn_rcpf(1.0f + __expf(-x));
}

// ---------------------------------------------------------------------------
// Weight pre-swizzle: f32 row-major [K=128][N] -> f16 WMMA-B fragment order.
// Fragment (nTile, kStep) is 1024 contiguous bytes; lane holds 16 halves.
// Lane mapping (16-bit operands, wave32):
//   lanes 0-15  : K_local {0..7, 16..23},  lanes 16-31 : K_local {8..15, 24..31}
//   halves 0..7 = K_local base..base+7, halves 8..15 = base+16..base+23
// ---------------------------------------------------------------------------
__global__ void asq_swizzle_w(const float* __restrict__ src, int N,
                              _Float16* __restrict__ dst) {
    int idx = blockIdx.x * blockDim.x + threadIdx.x;
    int total = 128 * N;
    if (idx >= total) return;
    int k = idx / N, n = idx % N;
    int kk = k & 31, kstep = k >> 5, nTile = n >> 4;
    int laneHigh = (kk >> 3) & 1;                 // kk in {8..15, 24..31}
    int e = (kk & 7) | ((kk & 16) >> 1);          // half index 0..15
    int lane = (n & 15) + (laneHigh ? 16 : 0);
    size_t o = ((size_t)(nTile * 4 + kstep)) * 512 + (size_t)lane * 16 + e;
    dst[o] = (_Float16)src[(size_t)k * N + n];
}

// A fragment (16x32 f16) from row-major LDS x; two aligned 16B chunks per lane.
__device__ __forceinline__ v16h asq_load_a(const _Float16* xv, int row0,
                                           int ks, int lane) {
    int m = row0 + (lane & 15);
    int kb = ks * 32 + ((lane < 16) ? 0 : 8);
    const _Float16* p = xv + m * asq::kXStride + kb;
    v8h lo = *(const v8h*)(p);
    v8h hi = *(const v8h*)(p + 16);
    v16h a;
#pragma unroll
    for (int i = 0; i < 8; ++i) { a[i] = lo[i]; a[i + 8] = hi[i]; }
    return a;
}

// B fragment: contiguous 32B per lane from pre-swizzled global weights.
__device__ __forceinline__ v16h asq_load_b(const _Float16* w, int nTile,
                                           int ks, int lane) {
    return ((const v16h*)w)[(nTile * 4 + ks) * 32 + lane];
}

__global__ __launch_bounds__(256)
void asq_attn_kernel(const int*   __restrict__ posid,
                     const float* __restrict__ qcv,
                     const float* __restrict__ mask,
                     const float* __restrict__ posembed,
                     const float* __restrict__ Wq,  const float* __restrict__ bq,
                     const float* __restrict__ Wqc, const float* __restrict__ bqc,
                     const float* __restrict__ bk,  const float* __restrict__ bkc,
                     const float* __restrict__ bv,  const float* __restrict__ bvc,
                     const float* __restrict__ lng, const float* __restrict__ lnb,
                     const _Float16* __restrict__ wswz,
                     float* __restrict__ out) {
    using namespace asq;
    extern __shared__ char smem[];
    _Float16* xv    = (_Float16*)(smem + kLdsXv);
    float*    k_lds = (float*)(smem + kLdsK);
    float*    probs = (float*)(smem + kLdsProbs);
    float*    q_sh  = (float*)(smem + kLdsQ);
    float*    red   = (float*)(smem + kLdsRed);    // [row][wave*2 + {sum,sq}]
    float*    murs  = (float*)(smem + kLdsMurs);   // [row]*2 -> {mu, rsig}

    const int b    = blockIdx.x;
    const int tid  = threadIdx.x;
    const int wave = tid >> 5;
    const int lane = tid & 31;

    // ---- 1) build x = concat(qcv, posembed[posid]) as f16 in LDS ----------
    for (int idx = tid; idx < kSPad * kInF; idx += 256) {
        int row = idx / kInF, c = idx % kInF;
        float val = 0.0f;
        if (row < kS) {
            if (c < kInQ) {
                val = qcv[((size_t)b * kS + row) * kInQ + c];
            } else {
                int p = posid[(size_t)b * kS + row];
                val = posembed[p * kPosE + (c - kInQ)];
            }
        }
        xv[row * kXStride + c] = (_Float16)val;
    }
    __syncthreads();

    // ---- 2) q = css(x[0]) on VALU (tiny: 128 outputs of 128-dot) ----------
    if (tid < 128) {
        float acc = bq[tid], accc = bqc[tid];
        for (int i = 0; i < kInF; ++i) {
            float xi = (float)xv[i];
            acc  += xi * Wq[i * 128 + tid];
            accc += xi * Wqc[i * 128 + tid];
        }
        q_sh[tid] = acc * asq_sigmoid(accc);
    }

    // ---- 3) K projection: wave w owns N-tile w (cols w*16..w*16+15) -------
    {
        const _Float16* wk  = wswz + kOffWk;
        const _Float16* wkc = wswz + kOffWkc;
        const int col = wave * 16 + (lane & 15);
        const float bkv = bk[col], bkcv = bkc[col];
        for (int m = 0; m < kMT; ++m) {
            v8f ck = {}; v8f ckc = {};
#pragma unroll
            for (int ks = 0; ks < 4; ++ks) {
                v16h a  = asq_load_a(xv, m * 16, ks, lane);
                v16h b1 = asq_load_b(wk,  wave, ks, lane);
                v16h b2 = asq_load_b(wkc, wave, ks, lane);
                ck  = __builtin_amdgcn_wmma_f32_16x16x32_f16(false, a, false, b1,
                                                             (short)0, ck, false, false);
                ckc = __builtin_amdgcn_wmma_f32_16x16x32_f16(false, a, false, b2,
                                                             (short)0, ckc, false, false);
            }
            const int rbase = m * 16 + ((lane < 16) ? 0 : 8);
#pragma unroll
            for (int r = 0; r < 8; ++r) {
                float kvv = (ck[r] + bkv) * asq_sigmoid(ckc[r] + bkcv);
                k_lds[(rbase + r) * kKStride + col] = kvv;
            }
        }
    }
    __syncthreads();

    // ---- 4) scores + softmax: wave h handles head h -----------------------
    {
        float qreg[kQLen];
#pragma unroll
        for (int d = 0; d < kQLen; ++d) qreg[d] = q_sh[wave * kQLen + d];
        float tv[7];
        float mx = -1e30f;
#pragma unroll
        for (int i = 0; i < 7; ++i) {
            int s = lane + 32 * i;
            float t = -1e30f;
            if (s < kS) {
                float acc = 0.0f;
#pragma unroll
                for (int d = 0; d < kQLen; ++d)
                    acc += qreg[d] * k_lds[s * kKStride + wave * kQLen + d];
                // mask added BEFORE the 1/sqrt(qlen) scaling, as in reference
                t = (acc + (1.0f - mask[s]) * (-10000.0f)) * 0.25f;
            }
            tv[i] = t;
            mx = fmaxf(mx, t);
        }
#pragma unroll
        for (int off = 16; off >= 1; off >>= 1)
            mx = fmaxf(mx, __shfl_xor(mx, off, 32));
        float sum = 0.0f;
#pragma unroll
        for (int i = 0; i < 7; ++i) {
            int s = lane + 32 * i;
            float e = (s < kS) ? __expf(tv[i] - mx) : 0.0f;
            tv[i] = e;
            sum += e;
        }
#pragma unroll
        for (int off = 16; off >= 1; off >>= 1)
            sum += __shfl_xor(sum, off, 32);
        float inv = __builtin_amdgcn_rcpf(sum);
#pragma unroll
        for (int i = 0; i < 7; ++i) {
            int s = lane + 32 * i;
            if (s < kSPad)
                probs[wave * kSPad + s] = (s < kS) ? tv[i] * inv : 0.0f;
        }
    }
    __syncthreads();

    // ---- 5) V projection + LayerNorm + ctx, streamed per 16-row M-tile ----
    // Wave h owns V columns h*64..h*64+63 == exactly head h's VLEN slice.
    const _Float16* wv  = wswz + kOffWv;
    const _Float16* wvc = wswz + kOffWvc;
    float ctx_acc[4] = {0.f, 0.f, 0.f, 0.f};
    float bvn[4], bvcn[4], gcol[4], bcol[4];
#pragma unroll
    for (int n = 0; n < 4; ++n) {
        int col = wave * kVLen + n * 16 + (lane & 15);
        bvn[n] = bv[col]; bvcn[n] = bvc[col];
        gcol[n] = lng[col]; bcol[n] = lnb[col];
    }

    for (int m = 0; m < kMT; ++m) {
        v8f cv[4] = {}; v8f cvc[4] = {};
#pragma unroll
        for (int ks = 0; ks < 4; ++ks) {
            v16h a = asq_load_a(xv, m * 16, ks, lane);
#pragma unroll
            for (int n = 0; n < 4; ++n) {
                int nTile = wave * 4 + n;
                v16h b1 = asq_load_b(wv,  nTile, ks, lane);
                v16h b2 = asq_load_b(wvc, nTile, ks, lane);
                cv[n]  = __builtin_amdgcn_wmma_f32_16x16x32_f16(false, a, false, b1,
                                                                (short)0, cv[n], false, false);
                cvc[n] = __builtin_amdgcn_wmma_f32_16x16x32_f16(false, a, false, b2,
                                                                (short)0, cvc[n], false, false);
            }
        }
        // gated V values + per-row LN partial sums (rows r / r+8 per lane half)
        float g[4][8];
        float s1[8], s2[8];
#pragma unroll
        for (int r = 0; r < 8; ++r) { s1[r] = 0.f; s2[r] = 0.f; }
#pragma unroll
        for (int n = 0; n < 4; ++n) {
#pragma unroll
            for (int r = 0; r < 8; ++r) {
                float val = (cv[n][r] + bvn[n]) * asq_sigmoid(cvc[n][r] + bvcn[n]);
                g[n][r] = val;
                s1[r] += val;
                s2[r] += val * val;
            }
        }
        // reduce across the 16 lanes of each half (xor masks < 16 stay in-half)
#pragma unroll
        for (int off = 1; off < 16; off <<= 1) {
#pragma unroll
            for (int r = 0; r < 8; ++r) {
                s1[r] += __shfl_xor(s1[r], off, 32);
                s2[r] += __shfl_xor(s2[r], off, 32);
            }
        }
        if ((lane & 15) == 0) {
            int rowOff = (lane < 16) ? 0 : 8;
#pragma unroll
            for (int r = 0; r < 8; ++r) {
                red[(rowOff + r) * 16 + wave * 2 + 0] = s1[r];
                red[(rowOff + r) * 16 + wave * 2 + 1] = s2[r];
            }
        }
        __syncthreads();
        if (tid < 16) {
            float su = 0.f, sq = 0.f;
#pragma unroll
            for (int w = 0; w < 8; ++w) {
                su += red[tid * 16 + w * 2 + 0];
                sq += red[tid * 16 + w * 2 + 1];
            }
            float mu  = su * (1.0f / (float)kHid);
            float var = sq * (1.0f / (float)kHid) - mu * mu;
            murs[tid * 2 + 0] = mu;
            murs[tid * 2 + 1] = __builtin_amdgcn_rsqf(var + kLnEps);
        }
        __syncthreads();
        // normalize and contract with probs
        const int rbase = (lane < 16) ? 0 : 8;
        float pr[8];
#pragma unroll
        for (int r = 0; r < 8; ++r)
            pr[r] = probs[wave * kSPad + m * 16 + rbase + r];
#pragma unroll
        for (int r = 0; r < 8; ++r) {
            float mu = murs[(rbase + r) * 2 + 0];
            float rs = murs[(rbase + r) * 2 + 1];
#pragma unroll
            for (int n = 0; n < 4; ++n) {
                float vn = (g[n][r] - mu) * rs * gcol[n] + bcol[n];
                ctx_acc[n] += pr[r] * vn;
            }
        }
    }

    // combine the two lane halves (rows 0-7 + rows 8-15) and write out
#pragma unroll
    for (int n = 0; n < 4; ++n)
        ctx_acc[n] += __shfl_xor(ctx_acc[n], 16, 32);
    if (lane < 16) {
#pragma unroll
        for (int n = 0; n < 4; ++n)
            out[(size_t)b * kHid + wave * kVLen + n * 16 + lane] = ctx_acc[n];
    }
}

extern "C" void kernel_launch(void* const* d_in, const int* in_sizes, int n_in,
                              void* d_out, int out_size, void* d_ws, size_t ws_size,
                              hipStream_t stream) {
    using namespace asq;
    const int*   posid    = (const int*)  d_in[0];
    const float* qcv      = (const float*)d_in[1];
    const float* mask     = (const float*)d_in[2];
    const float* posembed = (const float*)d_in[3];
    const float* Wq       = (const float*)d_in[4];
    const float* bq       = (const float*)d_in[5];
    const float* Wqc      = (const float*)d_in[6];
    const float* bqc      = (const float*)d_in[7];
    const float* Wk       = (const float*)d_in[8];
    const float* bk       = (const float*)d_in[9];
    const float* Wkc      = (const float*)d_in[10];
    const float* bkc      = (const float*)d_in[11];
    const float* Wv       = (const float*)d_in[12];
    const float* bv       = (const float*)d_in[13];
    const float* Wvc      = (const float*)d_in[14];
    const float* bvc      = (const float*)d_in[15];
    const float* lng      = (const float*)d_in[16];
    const float* lnb      = (const float*)d_in[17];

    _Float16* ws16 = (_Float16*)d_ws;   // needs >= 320 KB (weights as f16)

    // pre-swizzle weights into WMMA B-fragment order (L2-resident afterwards)
    asq_swizzle_w<<<(128 * 128 + 255) / 256, 256, 0, stream>>>(Wk,  128, ws16 + kOffWk);
    asq_swizzle_w<<<(128 * 128 + 255) / 256, 256, 0, stream>>>(Wkc, 128, ws16 + kOffWkc);
    asq_swizzle_w<<<(128 * 512 + 255) / 256, 256, 0, stream>>>(Wv,  512, ws16 + kOffWv);
    asq_swizzle_w<<<(128 * 512 + 255) / 256, 256, 0, stream>>>(Wvc, 512, ws16 + kOffWvc);

    asq_attn_kernel<<<kB, 256, kLdsBytes, stream>>>(
        posid, qcv, mask, posembed, Wq, bq, Wqc, bqc,
        bk, bkc, bv, bvc, lng, lnb, ws16, (float*)d_out);
}
